// OT_Attn_1073741824716
// MI455X (gfx1250) — matrix operand order repaired and checked by reading, not compiled, not run on hardware
//
#include <hip/hip_runtime.h>
#include <hip/hip_bf16.h>

#define NDIM 2048
#define DDIM 512
#define BATCH 4
#define NITER 250

typedef __attribute__((ext_vector_type(8)))  __bf16 v8bf;
typedef __attribute__((ext_vector_type(16))) __bf16 v16bf;
typedef __attribute__((ext_vector_type(8)))  float  v8f;
typedef __attribute__((ext_vector_type(4)))  int    v4i;
typedef __attribute__((address_space(1)))    v4i    v4i_glob;   // global 128-bit chunk
typedef __attribute__((address_space(3)))    v4i    v4i_lds;    // LDS 128-bit chunk

// Probe gfx1250 async global->LDS path (fallback to plain loads if absent)
#define OT_HAS_ASYNC_LDS 0
#if defined(__has_builtin)
#if __has_builtin(__builtin_amdgcn_global_load_async_to_lds_b128)
#undef OT_HAS_ASYNC_LDS
#define OT_HAS_ASYNC_LDS 1
#endif
#endif

__device__ inline unsigned short f2bf(float f) {
    unsigned u = __float_as_uint(f);
    unsigned r = u + 0x7FFFu + ((u >> 16) & 1u);   // round-to-nearest-even
    return (unsigned short)(r >> 16);
}
__device__ inline float bflo(unsigned p) { return __uint_as_float(p << 16); }
__device__ inline float bfhi(unsigned p) { return __uint_as_float(p & 0xffff0000u); }

// ---------------------------------------------------------------------------
// Init u, v to 1/n and per-batch cost max to 0
// ---------------------------------------------------------------------------
__global__ void ot_init_kernel(float* __restrict__ u, float* __restrict__ v,
                               unsigned int* __restrict__ cmax) {
    int idx = blockIdx.x * 256 + threadIdx.x;
    if (idx < BATCH * NDIM) {
        u[idx] = 1.0f / NDIM;
        v[idx] = 1.0f / NDIM;
    }
    if (idx < BATCH) cmax[idx] = 0u;
}

// ---------------------------------------------------------------------------
// Row-min normalize + bf16 convert + row sum-of-squares. One wave per row.
// Rows 0..8191 -> x, rows 8192..16383 -> y.
// ---------------------------------------------------------------------------
__global__ void ot_normalize_kernel(const float* __restrict__ x, const float* __restrict__ y,
                                    unsigned short* __restrict__ xbf, unsigned short* __restrict__ ybf,
                                    float* __restrict__ xsq, float* __restrict__ ysq) {
    int wave = threadIdx.x >> 5, lane = threadIdx.x & 31;
    int gw = blockIdx.x * 8 + wave;                    // 0..16383
    const int ROWS = BATCH * NDIM;
    const float* src; unsigned short* dstb; float* dsts; int row;
    if (gw < ROWS) { src = x; dstb = xbf; dsts = xsq; row = gw; }
    else           { src = y; dstb = ybf; dsts = ysq; row = gw - ROWS; }

    const float* rp = src + (size_t)row * DDIM;
    float vals[16];
    float mn = 3.4e38f;
#pragma unroll
    for (int k = 0; k < 16; ++k) { vals[k] = rp[lane + 32 * k]; mn = fminf(mn, vals[k]); }
#pragma unroll
    for (int m = 16; m >= 1; m >>= 1) mn = fminf(mn, __shfl_xor(mn, m, 32));

    float sq = 0.f;
    unsigned short* op = dstb + (size_t)row * DDIM;
#pragma unroll
    for (int k = 0; k < 16; ++k) {
        float vv = vals[k] - mn;
        sq += vv * vv;
        op[lane + 32 * k] = f2bf(vv);
    }
#pragma unroll
    for (int m = 16; m >= 1; m >>= 1) sq += __shfl_xor(sq, m, 32);
    if (lane == 0) dsts[row] = sq;
}

// ---------------------------------------------------------------------------
// Cost GEMM: C[b][i][j] = max(asq[i]+bsq[j]-2*(a_i . b_j), 0) (stored bf16)
// via v_wmma_f32_16x16x32_bf16; per-batch max via shuffle + uint atomicMax.
// One wave computes one 16x16 output tile; 8 waves/block.
// Called twice: (x,y)->Cxy and (y,x)->Cyx (the transpose).
// ---------------------------------------------------------------------------
__global__ void ot_cost_gemm_kernel(const unsigned short* __restrict__ abf,
                                    const unsigned short* __restrict__ bbf,
                                    const float* __restrict__ asq,
                                    const float* __restrict__ bsq,
                                    unsigned short* __restrict__ Cmat,
                                    unsigned int* __restrict__ cmax) {
    const int b    = blockIdx.y;
    const int wave = threadIdx.x >> 5;
    const int lane = threadIdx.x & 31;
    const int tile = blockIdx.x * 8 + wave;            // 0..16383
    const int ti   = (tile >> 7) << 4;                 // i-tile base
    const int tj   = (tile & 127) << 4;                // j-tile base

    const unsigned short* xb = abf + (size_t)b * NDIM * DDIM;
    const unsigned short* yb = bbf + (size_t)b * NDIM * DDIM;

    const int half = lane >> 4;                        // 0: lanes 0-15, 1: lanes 16-31
    const int l15  = lane & 15;
    const int arow = ti + l15;                         // A row loaded by this lane
    const int bcol = tj + l15;                         // B column loaded by this lane

    v8f c = {};
    for (int kk = 0; kk < DDIM; kk += 32) {
        // A 16x32 bf16: lane<16 holds K kk..kk+15; lane>=16: +16
        const unsigned short* ap = xb + (size_t)arow * DDIM + kk + half * 16;
        v8bf a0 = *(const v8bf*)(ap);
        v8bf a1 = *(const v8bf*)(ap + 8);
        v16bf a = __builtin_shufflevector(a0, a1, 0,1,2,3,4,5,6,7,8,9,10,11,12,13,14,15);
        // B 32x16 bf16 (B = b^T): lane<16 holds K kk..kk+15 of column bcol; lane>=16: +16
        const unsigned short* bp = yb + (size_t)bcol * DDIM + kk + half * 16;
        v16bf bm = *(const v16bf*)(bp);
        c = __builtin_amdgcn_wmma_f32_16x16x32_bf16(false, a, false, bm,
                                                    (short)0, c, false, false);
    }

    // C/D layout: lane<16 -> (M=r, N=lane); lane>=16 -> (M=r+8, N=lane-16)
    const float ysv = bsq[b * NDIM + bcol];
    unsigned short* Kb = Cmat + (size_t)b * NDIM * NDIM;
    float lmax = 0.f;
#pragma unroll
    for (int r = 0; r < 8; ++r) {
        int row = ti + r + half * 8;
        float cost = fmaxf(asq[b * NDIM + row] + ysv - 2.0f * c[r], 0.f);
        Kb[(size_t)row * NDIM + bcol] = f2bf(cost);
        lmax = fmaxf(lmax, cost);
    }
#pragma unroll
    for (int m = 16; m >= 1; m >>= 1) lmax = fmaxf(lmax, __shfl_xor(lmax, m, 32));
    if (lane == 0) atomicMax(&cmax[b], __float_as_uint(lmax));   // cost >= 0 -> order-safe
}

// ---------------------------------------------------------------------------
// K = exp(-cost/(reg*max)) in place over BOTH bf16 cost buffers (contiguous).
// Layout: 8 planes of 2^22 elems = 2^21 uints; plane&3 = batch.
// ---------------------------------------------------------------------------
__global__ void ot_expk_kernel(unsigned int* __restrict__ Kboth,
                               const unsigned int* __restrict__ cmax) {
    const float NEG_LOG2E_OVER_REG = -14.426950408889634f;   // -log2(e)/0.1
    size_t total  = (size_t)2 * BATCH * NDIM * NDIM / 2;     // uints
    size_t stride = (size_t)gridDim.x * blockDim.x;
    for (size_t idx = (size_t)blockIdx.x * blockDim.x + threadIdx.x; idx < total; idx += stride) {
        int b = (int)((idx >> 21) & 3);
        float s = NEG_LOG2E_OVER_REG / __uint_as_float(cmax[b]);
        unsigned p = Kboth[idx];
        float lo = exp2f(bflo(p) * s);
        float hi = exp2f(bfhi(p) * s);
        Kboth[idx] = (unsigned)f2bf(lo) | ((unsigned)f2bf(hi) << 16);
    }
}

// ---------------------------------------------------------------------------
// Shared Sinkhorn half-step: wout[r] = (1/N / sum_j M[r][j]*win[j])^fi
// M is bf16 row-major (2048x2048 per batch). One wave per row, win staged into
// LDS via async global->LDS DMA (ASYNCcnt) when available.
// Used as:  matvec(Kxy, v, u)  then  matvec(Kyx, u, v).
// ---------------------------------------------------------------------------
__global__ void ot_matvec_pow_kernel(const unsigned short* __restrict__ Mbf,
                                     const float* __restrict__ win,
                                     float* __restrict__ wout) {
    __shared__ float sw[NDIM];
    const int b = blockIdx.y;
    const float* wb = win + b * NDIM;

#if OT_HAS_ASYNC_LDS
    for (int base = threadIdx.x * 4; base < NDIM; base += 256 * 4) {
        __builtin_amdgcn_global_load_async_to_lds_b128(
            (v4i_glob*)(wb + base),
            (v4i_lds*)(sw + base),
            0, 0);
    }
#if __has_builtin(__builtin_amdgcn_s_wait_asynccnt)
    __builtin_amdgcn_s_wait_asynccnt(0);
#else
    asm volatile("s_wait_asynccnt 0" ::: "memory");
#endif
    __syncthreads();
#else
    for (int i = threadIdx.x; i < NDIM; i += 256) sw[i] = wb[i];
    __syncthreads();
#endif

    const int wave = threadIdx.x >> 5, lane = threadIdx.x & 31;
    const int row = blockIdx.x * 8 + wave;
    const unsigned* Kr = (const unsigned*)(Mbf + ((size_t)b * NDIM + row) * NDIM);
    float acc = 0.f;
#pragma unroll
    for (int s = 0; s < 8; ++s) {
        int j = lane * 8 + s * 256;                       // element index, 16B aligned
        uint4 p = *(const uint4*)(Kr + (j >> 1));
        acc += bflo(p.x) * sw[j + 0];
        acc += bfhi(p.x) * sw[j + 1];
        acc += bflo(p.y) * sw[j + 2];
        acc += bfhi(p.y) * sw[j + 3];
        acc += bflo(p.z) * sw[j + 4];
        acc += bfhi(p.z) * sw[j + 5];
        acc += bflo(p.w) * sw[j + 6];
        acc += bfhi(p.w) * sw[j + 7];
    }
#pragma unroll
    for (int m = 16; m >= 1; m >>= 1) acc += __shfl_xor(acc, m, 32);
    if (lane == 0) {
        const float a  = 1.0f / NDIM;
        const float fi = 0.8333333333333334f;             // reg_m/(reg_m+reg) = 0.5/0.6
        wout[b * NDIM + row] = exp2f(fi * log2f(a / acc));
    }
}

// ---------------------------------------------------------------------------
// out[b][j][i] = u[i] * KT[b][j][i] * v[j]; KT row-major over (j,i) so both
// reads and float4 writes are coalesced. 8 elements per thread.
// ---------------------------------------------------------------------------
__global__ void ot_out_kernel(const unsigned short* __restrict__ KT,
                              const float* __restrict__ u,
                              const float* __restrict__ v,
                              float* __restrict__ out) {
    size_t t = (size_t)blockIdx.x * 256 + threadIdx.x;
    size_t base = t * 8;                                  // element index (b,j,i) row-major
    int b = (int)(base >> 22);
    int rem = (int)(base & ((1u << 22) - 1));
    int j = rem >> 11, i0 = rem & (NDIM - 1);

    uint4 p = *(const uint4*)(KT + base);
    float vj = v[b * NDIM + j];
    const float* ub = u + b * NDIM + i0;
    float4 o0, o1;
    o0.x = vj * ub[0] * bflo(p.x);
    o0.y = vj * ub[1] * bfhi(p.x);
    o0.z = vj * ub[2] * bflo(p.y);
    o0.w = vj * ub[3] * bfhi(p.y);
    o1.x = vj * ub[4] * bflo(p.z);
    o1.y = vj * ub[5] * bfhi(p.z);
    o1.z = vj * ub[6] * bflo(p.w);
    o1.w = vj * ub[7] * bfhi(p.w);
    float4* op = (float4*)(out + base);
    op[0] = o0;
    op[1] = o1;
}

// ---------------------------------------------------------------------------
extern "C" void kernel_launch(void* const* d_in, const int* in_sizes, int n_in,
                              void* d_out, int out_size, void* d_ws, size_t ws_size,
                              hipStream_t stream) {
    const float* x = (const float*)d_in[0];
    const float* y = (const float*)d_in[1];
    float* out = (float*)d_out;

    // Workspace layout (~81 MB). Cxy and Cyx are adjacent so expk sweeps both.
    char* ws = (char*)d_ws;
    size_t off = 0;
    unsigned short* Cxy = (unsigned short*)(ws + off); off += (size_t)BATCH * NDIM * NDIM * 2; // 32 MB
    unsigned short* Cyx = (unsigned short*)(ws + off); off += (size_t)BATCH * NDIM * NDIM * 2; // 32 MB
    unsigned short* xbf = (unsigned short*)(ws + off); off += (size_t)BATCH * NDIM * DDIM * 2; // 8 MB
    unsigned short* ybf = (unsigned short*)(ws + off); off += (size_t)BATCH * NDIM * DDIM * 2; // 8 MB
    float* xsq = (float*)(ws + off);                   off += (size_t)BATCH * NDIM * 4;
    float* ysq = (float*)(ws + off);                   off += (size_t)BATCH * NDIM * 4;
    float* u   = (float*)(ws + off);                   off += (size_t)BATCH * NDIM * 4;
    float* v   = (float*)(ws + off);                   off += (size_t)BATCH * NDIM * 4;
    unsigned int* cmax = (unsigned int*)(ws + off);    off += 256;
    (void)ws_size; (void)in_sizes; (void)n_in; (void)out_size;

    ot_init_kernel<<<32, 256, 0, stream>>>(u, v, cmax);
    ot_normalize_kernel<<<2048, 256, 0, stream>>>(x, y, xbf, ybf, xsq, ysq);
    // Cxy = cost(x,y); Cyx = cost(y,x) == cost(x,y)^T  (max is shared, idempotent)
    ot_cost_gemm_kernel<<<dim3(2048, BATCH), 256, 0, stream>>>(xbf, ybf, xsq, ysq, Cxy, cmax);
    ot_cost_gemm_kernel<<<dim3(2048, BATCH), 256, 0, stream>>>(ybf, xbf, ysq, xsq, Cyx, cmax);
    ot_expk_kernel<<<8192, 256, 0, stream>>>((unsigned int*)Cxy, cmax);  // both buffers

    for (int it = 0; it < NITER; ++it) {
        ot_matvec_pow_kernel<<<dim3(256, BATCH), 256, 0, stream>>>(Cxy, v, u); // u=(a/Kv)^fi
        ot_matvec_pow_kernel<<<dim3(256, BATCH), 256, 0, stream>>>(Cyx, u, v); // v=(b/K^T u)^fi
    }

    ot_out_kernel<<<8192, 256, 0, stream>>>(Cyx, u, v, out);
}
